// CausalAttention_65068754534447
// MI455X (gfx1250) — matrix operand order repaired and checked by reading, not compiled, not run on hardware
//
#include <hip/hip_runtime.h>
#include <hip/hip_bf16.h>

typedef __bf16 bf16_t;
typedef bf16_t v4bf  __attribute__((ext_vector_type(4)));
typedef bf16_t v8bf  __attribute__((ext_vector_type(8)));
typedef bf16_t v16bf __attribute__((ext_vector_type(16)));
typedef float  v8f   __attribute__((ext_vector_type(8)));

constexpr int kB  = 4;
constexpr int kS  = 2048;
constexpr int kD  = 1024;
constexpr int kH  = 16;
constexpr int kDH = 64;
constexpr int kM  = kB * kS;   // 8192 rows for the QKV GEMM

__device__ __forceinline__ v16bf cat8(v8bf lo, v8bf hi) {
  return __builtin_shufflevector(lo, hi, 0,1,2,3,4,5,6,7,8,9,10,11,12,13,14,15);
}

__device__ __forceinline__ v8f wmma_bf16(v16bf a, v16bf b, v8f c) {
  // D = A(16x32 bf16) * B(32x16 bf16) + C(16x16 f32)
  return __builtin_amdgcn_wmma_f32_16x16x32_bf16(false, a, false, b, (short)0, c,
                                                 false, false);
}

__device__ __forceinline__ v8f v8f_zero() {
  v8f z;
#pragma unroll
  for (int i = 0; i < 8; ++i) z[i] = 0.0f;
  return z;
}

// CDNA5 async copy: global -> LDS, 16 bytes per lane, tracked by ASYNCcnt.
// GV addressing: VDST = per-lane LDS byte address, VADDR = 64-bit global addr.
__device__ __forceinline__ void async_load_b128_to_lds(const bf16_t* gptr,
                                                       bf16_t* lptr) {
  __attribute__((address_space(3))) bf16_t* lp =
      (__attribute__((address_space(3))) bf16_t*)lptr;
  unsigned long long ga = (unsigned long long)(uintptr_t)gptr;
  asm volatile("global_load_async_to_lds_b128 %0, %1, off"
               :: "v"(lp), "v"(ga)
               : "memory");
}

__device__ __forceinline__ void wait_asynccnt0() {
  asm volatile("s_wait_asynccnt 0" ::: "memory");
}

// -----------------------------------------------------------------------------
// Y = X * W^T + bias  (X:[8192,1024] f32, W:[1024,1024] f32 row-major)
// Output bf16, reordered to [B, H, S, DH] head-major for the attention kernel.
// Block: 256 threads (8 waves), tile 128(M) x 64(N), K stepped by 64
// -> 8 WMMAs per barrier pair.
// -----------------------------------------------------------------------------
__global__ __launch_bounds__(256, 2)
void qkv_proj_kernel(const float* __restrict__ X, const float* __restrict__ W,
                     const float* __restrict__ bias, bf16_t* __restrict__ Y)
{
  __shared__ bf16_t As[128][72];   // 128x64 tile, rows padded to 72 (144B, 16B-mult)
  __shared__ bf16_t Bs[64][72];    // 64x64 tile of W rows (output features)

  const int tid  = threadIdx.x;
  const int lane = tid & 31;
  const int w    = tid >> 5;
  const int lrow = lane & 15;
  const int half = lane >> 4;
  const int wm   = w & 3;          // 4 waves down M  -> 32 rows each
  const int wn   = w >> 2;         // 2 waves across N -> 32 cols each
  const int mBase = blockIdx.x * 128;
  const int nBase = blockIdx.y * 64;

  v8f acc[2][2];
#pragma unroll
  for (int i = 0; i < 2; ++i)
#pragma unroll
    for (int j = 0; j < 2; ++j) acc[i][j] = v8f_zero();

  for (int k0 = 0; k0 < kD; k0 += 64) {
    // Stage A (128x64 f32 -> bf16): 2048 float4 chunks, 8 per thread.
#pragma unroll
    for (int j = 0; j < 8; ++j) {
      const int idx = tid + j * 256;
      const int r = idx >> 4, c = (idx & 15) * 4;
      float4 v = *reinterpret_cast<const float4*>(X + (size_t)(mBase + r) * kD + k0 + c);
      v4bf bv; bv[0] = (bf16_t)v.x; bv[1] = (bf16_t)v.y;
      bv[2] = (bf16_t)v.z; bv[3] = (bf16_t)v.w;
      *reinterpret_cast<v4bf*>(&As[r][c]) = bv;
    }
    // Stage B (64x64 f32 -> bf16): 1024 float4 chunks, 4 per thread.
#pragma unroll
    for (int j = 0; j < 4; ++j) {
      const int idx = tid + j * 256;
      const int r = idx >> 4, c = (idx & 15) * 4;
      float4 v = *reinterpret_cast<const float4*>(W + (size_t)(nBase + r) * kD + k0 + c);
      v4bf bv; bv[0] = (bf16_t)v.x; bv[1] = (bf16_t)v.y;
      bv[2] = (bf16_t)v.z; bv[3] = (bf16_t)v.w;
      *reinterpret_cast<v4bf*>(&Bs[r][c]) = bv;
    }
    __syncthreads();

#pragma unroll
    for (int kk = 0; kk < 2; ++kk) {
      // A fragment: lane holds row lrow; elems 0-7 -> K = half*8 + j,
      // elems 8-15 -> K = 16 + half*8 + j   (ISA 16-bit A 16x32 layout)
      v16bf afrag[2], bfrag[2];
#pragma unroll
      for (int mi = 0; mi < 2; ++mi) {
        const int row = wm * 32 + mi * 16 + lrow;
        v8bf a0 = *reinterpret_cast<const v8bf*>(&As[row][kk * 32 + half * 8]);
        v8bf a1 = *reinterpret_cast<const v8bf*>(&As[row][kk * 32 + 16 + half * 8]);
        afrag[mi] = cat8(a0, a1);
      }
      // B fragment: lane holds col lrow; elems 0-7 -> K = half*8 + j, elems 8-15 -> K = 16 + half*8 + j
#pragma unroll
      for (int ni = 0; ni < 2; ++ni) {
        const int col = wn * 32 + ni * 16 + lrow;
        v8bf b0 = *reinterpret_cast<const v8bf*>(&Bs[col][kk * 32 + half * 8]);
        v8bf b1 = *reinterpret_cast<const v8bf*>(&Bs[col][kk * 32 + 16 + half * 8]);
        bfrag[ni] = cat8(b0, b1);
      }
#pragma unroll
      for (int mi = 0; mi < 2; ++mi)
#pragma unroll
        for (int ni = 0; ni < 2; ++ni)
          acc[mi][ni] = wmma_bf16(afrag[mi], bfrag[ni], acc[mi][ni]);
    }
    __syncthreads();
  }

  // Epilogue: add bias, store bf16 into [B,H,S,DH] layout.
  // C layout: VGPR r holds row (r + 8*half), lane%16 is the column.
#pragma unroll
  for (int mi = 0; mi < 2; ++mi)
#pragma unroll
    for (int ni = 0; ni < 2; ++ni) {
      const int n  = nBase + wn * 32 + ni * 16 + lrow;
      const float bval = bias[n];
      const int h = n >> 6, dh = n & (kDH - 1);
#pragma unroll
      for (int r = 0; r < 8; ++r) {
        const int m  = mBase + wm * 32 + mi * 16 + r + 8 * half;
        const int bb = m >> 11, s = m & (kS - 1);
        const float val = acc[mi][ni][r] + bval;
        Y[(((size_t)bb * kH + h) * kS + s) * kDH + dh] = (bf16_t)val;
      }
    }
}

// -----------------------------------------------------------------------------
// Flash-style causal attention over bf16 Q/K/V in [B,H,S,DH].
// Block = 256 threads (8 waves), 128 query rows per block (16 per wave).
// Key loop in tiles of 32; K tile staged via async global->LDS copy (ASYNCcnt),
// V tile transposed manually; online softmax; f32 accumulators; fp32 output.
// -----------------------------------------------------------------------------
__global__ __launch_bounds__(256, 2)
void attn_kernel(const bf16_t* __restrict__ Q, const bf16_t* __restrict__ K,
                 const bf16_t* __restrict__ V, float* __restrict__ out)
{
  __shared__ bf16_t Ks[32][72];       // keys x dh (row-major, padded)
  __shared__ bf16_t Vt[64][40];       // dh x keys (transposed, padded)
  __shared__ bf16_t Ps[8][16][40];    // per-wave P staging (C-layout -> A-layout)

  const int tid  = threadIdx.x;
  const int lane = tid & 31;
  const int w    = tid >> 5;
  const int lrow = lane & 15;
  const int half = lane >> 4;
  const int bh   = blockIdx.y;
  const int bb   = bh >> 4;           // H = 16
  const int h    = bh & 15;
  const int qBase = blockIdx.x * 128;
  const int qRow0 = qBase + w * 16;

  const bf16_t* Qh = Q + (size_t)bh * kS * kDH;
  const bf16_t* Kh = K + (size_t)bh * kS * kDH;
  const bf16_t* Vh = V + (size_t)bh * kS * kDH;

  // Q fragments (A layout), loaded once: two K-steps of 32 over DH=64.
  v16bf qa[2];
  {
    const bf16_t* qp = Qh + (size_t)(qRow0 + lrow) * kDH;
#pragma unroll
    for (int kk = 0; kk < 2; ++kk) {
      v8bf a0 = *reinterpret_cast<const v8bf*>(qp + kk * 32 + half * 8);
      v8bf a1 = *reinterpret_cast<const v8bf*>(qp + kk * 32 + 16 + half * 8);
      qa[kk] = cat8(a0, a1);
    }
  }

  float mrun[8], lrun[8];
#pragma unroll
  for (int r = 0; r < 8; ++r) { mrun[r] = -1e30f; lrun[r] = 0.0f; }
  v8f oacc[4];
#pragma unroll
  for (int nt = 0; nt < 4; ++nt) oacc[nt] = v8f_zero();

  const int nkt = qBase / 32 + 4;     // causal bound for this query block
  for (int kt = 0; kt < nkt; ++kt) {
    const int kb = kt * 32;
    // Cooperative stage: K tile via CDNA5 async copy; V tile transposed by hand.
    {
      const int r = tid >> 3, c8 = (tid & 7) * 8;
      async_load_b128_to_lds(Kh + (size_t)(kb + r) * kDH + c8, &Ks[r][c8]);
      v8bf vv = *reinterpret_cast<const v8bf*>(Vh + (size_t)(kb + r) * kDH + c8);
#pragma unroll
      for (int i = 0; i < 8; ++i) Vt[c8 + i][r] = vv[i];
    }
    wait_asynccnt0();
    __syncthreads();

    const bool active = (kb <= qRow0 + 15);   // wave-uniform; EXEC stays all-1s
    if (active) {
      // scores: 2 key n-tiles x 2 K-steps
      v8f sc[2];
#pragma unroll
      for (int t = 0; t < 2; ++t) {
        sc[t] = v8f_zero();
#pragma unroll
        for (int kk = 0; kk < 2; ++kk) {
          v8bf b0 = *reinterpret_cast<const v8bf*>(&Ks[t * 16 + lrow][kk * 32 + half * 8]);
          v8bf b1 = *reinterpret_cast<const v8bf*>(&Ks[t * 16 + lrow][kk * 32 + 16 + half * 8]);
          sc[t] = wmma_bf16(qa[kk], cat8(b0, b1), sc[t]);
        }
      }

      // causal mask + online softmax (rows live in VGPR index + lane-half)
      float alpha[8];
#pragma unroll
      for (int r = 0; r < 8; ++r) {
        const int q = qRow0 + r + 8 * half;
        float s0 = sc[0][r] * 0.125f;           // 1/sqrt(64)
        float s1 = sc[1][r] * 0.125f;
        const bool msk0 = (kb + lrow) > q;
        const bool msk1 = (kb + 16 + lrow) > q;
        s0 = msk0 ? -1e30f : s0;
        s1 = msk1 ? -1e30f : s1;
        float v = fmaxf(s0, s1);
        v = fmaxf(v, __shfl_xor(v, 1, 32));
        v = fmaxf(v, __shfl_xor(v, 2, 32));
        v = fmaxf(v, __shfl_xor(v, 4, 32));
        v = fmaxf(v, __shfl_xor(v, 8, 32));
        const float mnew = fmaxf(mrun[r], v);
        const float a_   = __expf(mrun[r] - mnew);
        mrun[r] = mnew;
        alpha[r] = a_;
        const float p0 = msk0 ? 0.0f : __expf(s0 - mnew);
        const float p1 = msk1 ? 0.0f : __expf(s1 - mnew);
        float ps = p0 + p1;
        ps += __shfl_xor(ps, 1, 32);
        ps += __shfl_xor(ps, 2, 32);
        ps += __shfl_xor(ps, 4, 32);
        ps += __shfl_xor(ps, 8, 32);
        lrun[r] = lrun[r] * a_ + ps;
        Ps[w][r + 8 * half][lrow]      = (bf16_t)p0;
        Ps[w][r + 8 * half][16 + lrow] = (bf16_t)p1;
      }
#pragma unroll
      for (int nt = 0; nt < 4; ++nt)
#pragma unroll
        for (int r = 0; r < 8; ++r) oacc[nt][r] *= alpha[r];

      // per-wave LDS round trip (C-layout -> A-layout); order via DScnt
      asm volatile("s_wait_dscnt 0" ::: "memory");

      v8bf p0v = *reinterpret_cast<const v8bf*>(&Ps[w][lrow][half * 8]);
      v8bf p1v = *reinterpret_cast<const v8bf*>(&Ps[w][lrow][16 + half * 8]);
      const v16bf pa = cat8(p0v, p1v);
#pragma unroll
      for (int nt = 0; nt < 4; ++nt) {
        v8bf b0 = *reinterpret_cast<const v8bf*>(&Vt[nt * 16 + lrow][half * 8]);
        v8bf b1 = *reinterpret_cast<const v8bf*>(&Vt[nt * 16 + lrow][16 + half * 8]);
        oacc[nt] = wmma_bf16(pa, cat8(b0, b1), oacc[nt]);
      }
    }
    __syncthreads();
  }

  // Epilogue: normalize, store fp32 [B,S,D] with head interleave.
#pragma unroll
  for (int nt = 0; nt < 4; ++nt) {
    const int dh = nt * 16 + lrow;
#pragma unroll
    for (int r = 0; r < 8; ++r) {
      const int q = qRow0 + r + 8 * half;
      out[((size_t)(bb * kS + q)) * kD + h * kDH + dh] = oacc[nt][r] / lrun[r];
    }
  }
}

// -----------------------------------------------------------------------------
extern "C" void kernel_launch(void* const* d_in, const int* in_sizes, int n_in,
                              void* d_out, int out_size, void* d_ws, size_t ws_size,
                              hipStream_t stream)
{
  (void)in_sizes; (void)n_in; (void)out_size; (void)ws_size;

  const float* x  = (const float*)d_in[0];
  // d_in[1] is the causal mask; causality is applied analytically.
  const float* Wq = (const float*)d_in[2];
  const float* bq = (const float*)d_in[3];
  const float* Wk = (const float*)d_in[4];
  const float* bk = (const float*)d_in[5];
  const float* Wv = (const float*)d_in[6];
  const float* bv = (const float*)d_in[7];
  float* out = (float*)d_out;

  const size_t per = (size_t)kB * kH * kS * kDH;   // 8Mi bf16 elems each
  bf16_t* qws = (bf16_t*)d_ws;
  bf16_t* kws = qws + per;
  bf16_t* vws = kws + per;

  dim3 gProj(kM / 128, kD / 64);                    // 64 x 16 blocks
  qkv_proj_kernel<<<gProj, 256, 0, stream>>>(x, Wq, bq, qws);
  qkv_proj_kernel<<<gProj, 256, 0, stream>>>(x, Wk, bk, kws);
  qkv_proj_kernel<<<gProj, 256, 0, stream>>>(x, Wv, bv, vws);

  dim3 gAttn(kS / 128, kB * kH);                    // 16 x 64 blocks
  attn_kernel<<<gAttn, 256, 0, stream>>>(qws, kws, vws, out);
}
